// Sent_Posit_Drmm_Modeler_25598005084716
// MI455X (gfx1250) — compile-verified, matrix operand order checked by reading, not run on hardware
//
#include <hip/hip_runtime.h>
#include <stdint.h>

#define E_DIM 300
#define F_DIM 300
#define FSZ   3
#define N_SENT 1024
#define S_LEN 128
#define Q_LEN 32
#define EP    320   // zero-padded K extent (multiple of 32)
#define FP_   304   // padded filter-row count
#define XST   304   // LDS row stride in elements (multiple of 8 -> 16B aligned rows)

typedef __attribute__((ext_vector_type(16))) __bf16 v16bf;
typedef __attribute__((ext_vector_type(8)))  float  v8f;

union FragBF { v16bf v; uint4 u[2]; };

__device__ __forceinline__ unsigned short f2bf(float f) {
  unsigned u = __float_as_uint(f);
  unsigned r = u + 0x7FFFu + ((u >> 16) & 1u);   // round-to-nearest-even
  return (unsigned short)(r >> 16);
}
__device__ __forceinline__ float bf2f(unsigned short h) {
  return __uint_as_float(((unsigned)h) << 16);
}

// ---------------- prep: filters -> bf16, layout [FP_][FSZ][EP], zero padded -----
__global__ void prep_filters(const float* __restrict__ filters,
                             unsigned short* __restrict__ Wb) {
  int idx = blockIdx.x * blockDim.x + threadIdx.x;
  if (idx >= FP_ * FSZ * EP) return;
  int f = idx / (FSZ * EP);
  int r = idx - f * (FSZ * EP);
  int k = r / EP;
  int e = r - k * EP;
  float v = 0.f;
  if (f < F_DIM && e < E_DIM) v = filters[((size_t)f * FSZ + k) * E_DIM + e];
  Wb[idx] = f2bf(v);
}

// ---------------- prep: question embed + conv + norms ---------------------------
__global__ void prep_question(const float* __restrict__ embeds,
                              const float* __restrict__ filters,
                              const int*   __restrict__ question,
                              unsigned short* __restrict__ qbE,   // [Q_LEN][EP] bf16
                              unsigned short* __restrict__ qbC,   // [Q_LEN][EP] bf16
                              float* __restrict__ qnE,
                              float* __restrict__ qnC) {
  __shared__ float qel[Q_LEN * E_DIM];
  __shared__ float qc2[Q_LEN];
  int tid = threadIdx.x;
  if (tid < Q_LEN) qc2[tid] = 0.f;
  for (int idx = tid; idx < Q_LEN * E_DIM; idx += blockDim.x) {
    int q = idx / E_DIM, e = idx - q * E_DIM;
    qel[idx] = embeds[(size_t)question[q] * E_DIM + e];
  }
  __syncthreads();
  if (tid < Q_LEN) {
    float s = 0.f;
    for (int e = 0; e < E_DIM; ++e) { float v = qel[tid * E_DIM + e]; s += v * v; }
    qnE[tid] = sqrtf(s);
  }
  for (int idx = tid; idx < Q_LEN * EP; idx += blockDim.x) {
    int q = idx / EP, e = idx - q * EP;
    qbE[idx] = f2bf(e < E_DIM ? qel[q * E_DIM + e] : 0.f);
  }
  for (int idx = tid; idx < Q_LEN * F_DIM; idx += blockDim.x) {
    int q = idx / F_DIM, f = idx - q * F_DIM;
    float acc = 0.f;
    for (int k = 0; k < FSZ; ++k) {
      if (q + k < Q_LEN) {
        const float* xr = &qel[(q + k) * E_DIM];
        const float* wr = &filters[((size_t)f * FSZ + k) * E_DIM];
        for (int e = 0; e < E_DIM; ++e) acc += xr[e] * wr[e];
      }
    }
    qbC[q * EP + f] = f2bf(acc);
    atomicAdd(&qc2[q], acc * acc);
  }
  for (int idx = tid; idx < Q_LEN * EP; idx += blockDim.x) {
    int q = idx / EP, e = idx - q * EP;
    if (e >= F_DIM) qbC[idx] = 0;
  }
  __syncthreads();
  if (tid < Q_LEN) qnC[tid] = sqrtf(qc2[tid]);
}

// ---------------- main: one workgroup per (sentence, doc) -----------------------
__global__ __launch_bounds__(256)
void doc_kernel(const float* __restrict__ embeds,
                const int*   __restrict__ question,
                const int*   __restrict__ doc1,
                const int*   __restrict__ doc2,
                const float* __restrict__ sim1,
                const float* __restrict__ sim2,
                const unsigned short* __restrict__ Wb,
                const unsigned short* __restrict__ qbE,
                const unsigned short* __restrict__ qbC,
                const float* __restrict__ qnEg,
                const float* __restrict__ qnCg,
                const float* __restrict__ lin_w,
                const float* __restrict__ lin_b,
                float* __restrict__ out)
{
  __shared__ __align__(16) unsigned short Xe[131 * XST];  // emb bf16 (+2 zero rows, +1 overread row)
  __shared__ __align__(16) unsigned short Cc[129 * XST];  // conv bf16 (+1 overread row)
  __shared__ float nrmS[S_LEN], nrmC[S_LEN];
  __shared__ int   stok[S_LEN];
  __shared__ float qnE[Q_LEN], qnC[Q_LEN];
  __shared__ int   qidm[Q_LEN];
  __shared__ float featb[Q_LEN * 6];
  __shared__ float lo[Q_LEN];

  const int n      = blockIdx.x;
  const int docSel = blockIdx.y;
  const int*   dids = docSel ? doc2 : doc1;
  const float* dsim = docSel ? sim2 : sim1;

  const int tid  = threadIdx.x;
  const int w    = tid >> 5;
  const int lane = tid & 31;
  const int col  = lane & 15;
  const int hi   = lane >> 4;
  const int off  = hi * 8;   // K half-offset per A/B 16-bit WMMA lane layout

  if (tid < Q_LEN) { qnE[tid] = qnEg[tid]; qnC[tid] = qnCg[tid]; qidm[tid] = question[tid]; }
  for (int idx = tid; idx < 3 * XST; idx += 256) Xe[128 * XST + idx] = 0;

  // ---- Phase A: gather sentence embeddings, fp32 norms, bf16 to LDS ----
  for (int i = 0; i < 16; ++i) {
    int r = 16 * w + i;
    int tok = dids[n * S_LEN + r];
    if (lane == 0) stok[r] = tok;
    const float* row = embeds + (size_t)tok * E_DIM;
    float ss = 0.f;
    for (int c = 0; c < 10; ++c) {
      int e = lane + 32 * c;
      if (e < E_DIM) { float v = row[e]; Xe[r * XST + e] = f2bf(v); ss += v * v; }
      else if (e < XST) Xe[r * XST + e] = 0;
    }
    for (int m = 16; m >= 1; m >>= 1) ss += __shfl_xor(ss, m, 32);
    if (lane == 0) nrmS[r] = sqrtf(ss);
  }
  __syncthreads();

  // ---- Phase B: causal-window conv as GEMM via WMMA bf16 ----
  for (int nt = 0; nt < 19; ++nt) {
    v8f acc = {0.f, 0.f, 0.f, 0.f, 0.f, 0.f, 0.f, 0.f};
    int f = nt * 16 + col;
    for (int k = 0; k < FSZ; ++k) {
      const unsigned short* wrow = Wb + ((size_t)f * FSZ + k) * EP;
      const unsigned short* xrow = &Xe[(16 * w + col + k) * XST];
#pragma unroll
      for (int ks = 0; ks < 10; ++ks) {
        int kb = ks * 32 + off;
        FragBF a, b;
        const uint4* pa = reinterpret_cast<const uint4*>(xrow + kb);
        a.u[0] = pa[0]; a.u[1] = pa[2];
        const uint4* pb = reinterpret_cast<const uint4*>(wrow + kb);
        b.u[0] = pb[0]; b.u[1] = pb[2];
        acc = __builtin_amdgcn_wmma_f32_16x16x32_bf16(false, a.v, false, b.v,
                                                      (short)0, acc, false, false);
      }
    }
#pragma unroll
    for (int v = 0; v < 8; ++v)
      Cc[(16 * w + v + 8 * hi) * XST + nt * 16 + col] = f2bf(acc[v]);
  }
  __syncthreads();

  // ---- Phase C: conv row norms ----
  for (int i = 0; i < 16; ++i) {
    int r = 16 * w + i;
    float ss = 0.f;
    for (int c = 0; c < 10; ++c) {
      int e = lane + 32 * c;
      if (e < F_DIM) { float v = bf2f(Cc[r * XST + e]); ss += v * v; }
    }
    for (int m = 16; m >= 1; m >>= 1) ss += __shfl_xor(ss, m, 32);
    if (lane == 0) nrmC[r] = sqrtf(ss);
  }
  __syncthreads();

  // ---- Phase D: cosine-sim GEMMs (insens: Xe x qbE, sens: Cc x qbC) ----
  v8f aI0 = {0.f,0.f,0.f,0.f,0.f,0.f,0.f,0.f};
  v8f aI1 = aI0, aS0 = aI0, aS1 = aI0;
  {
    const unsigned short* xrow = &Xe[(16 * w + col) * XST];
    const unsigned short* crow = &Cc[(16 * w + col) * XST];
#pragma unroll
    for (int ks = 0; ks < 10; ++ks) {
      int kb = ks * 32 + off;
      FragBF ae, ac, b0, b1, c0, c1;
      const uint4* p;
      p = reinterpret_cast<const uint4*>(xrow + kb);               ae.u[0] = p[0]; ae.u[1] = p[2];
      p = reinterpret_cast<const uint4*>(crow + kb);               ac.u[0] = p[0]; ac.u[1] = p[2];
      p = reinterpret_cast<const uint4*>(qbE + col * EP + kb);     b0.u[0] = p[0]; b0.u[1] = p[2];
      p = reinterpret_cast<const uint4*>(qbE + (16 + col) * EP + kb); b1.u[0] = p[0]; b1.u[1] = p[2];
      p = reinterpret_cast<const uint4*>(qbC + col * EP + kb);     c0.u[0] = p[0]; c0.u[1] = p[2];
      p = reinterpret_cast<const uint4*>(qbC + (16 + col) * EP + kb); c1.u[0] = p[0]; c1.u[1] = p[2];
      aI0 = __builtin_amdgcn_wmma_f32_16x16x32_bf16(false, ae.v, false, b0.v, (short)0, aI0, false, false);
      aI1 = __builtin_amdgcn_wmma_f32_16x16x32_bf16(false, ae.v, false, b1.v, (short)0, aI1, false, false);
      aS0 = __builtin_amdgcn_wmma_f32_16x16x32_bf16(false, ac.v, false, c0.v, (short)0, aS0, false, false);
      aS1 = __builtin_amdgcn_wmma_f32_16x16x32_bf16(false, ac.v, false, c1.v, (short)0, aS1, false, false);
    }
  }
  __syncthreads();   // all WMMA reads of Xe/Cc complete -> safe to overlay sims on Xe

  float* simI = reinterpret_cast<float*>(Xe);          // [S_LEN][Q_LEN] fp32
  float* simS = simI + S_LEN * Q_LEN;                  // [S_LEN][Q_LEN] fp32
#pragma unroll
  for (int v = 0; v < 8; ++v) {
    int row = 16 * w + v + 8 * hi;
    float ns = nrmS[row], nc = nrmC[row];
    int   sm = stok[row] > 1;
    float m0 = (sm && qidm[col] > 1)      ? 1.f : 0.f;
    float m1 = (sm && qidm[16 + col] > 1) ? 1.f : 0.f;
    simI[row * Q_LEN + col]      = aI0[v] / (qnE[col] * ns) * m0;
    simI[row * Q_LEN + 16 + col] = aI1[v] / (qnE[16 + col] * ns) * m1;
    simS[row * Q_LEN + col]      = aS0[v] / (qnC[col] * nc);
    simS[row * Q_LEN + 16 + col] = aS1[v] / (qnC[16 + col] * nc);
  }
  __syncthreads();

  // ---- Phase E: top-5 pooling over s (3 sources x 32 queries), then score ----
  if (tid < 96) {
    int src = tid >> 5;
    int q   = tid & 31;
    float t0 = -1e30f, t1 = -1e30f, t2 = -1e30f, t3 = -1e30f, t4 = -1e30f;
    for (int s = 0; s < S_LEN; ++s) {
      float v;
      if (src == 0)      v = simI[s * Q_LEN + q];
      else if (src == 1) v = simS[s * Q_LEN + q];
      else               v = dsim[((size_t)n * Q_LEN + q) * S_LEN + s];
      if (v > t4) {
        if (v > t0)      { t4 = t3; t3 = t2; t2 = t1; t1 = t0; t0 = v; }
        else if (v > t1) { t4 = t3; t3 = t2; t2 = t1; t1 = v; }
        else if (v > t2) { t4 = t3; t3 = t2; t2 = v; }
        else if (v > t3) { t4 = t3; t3 = v; }
        else             t4 = v;
      }
    }
    featb[q * 6 + src * 2]     = t0;
    featb[q * 6 + src * 2 + 1] = (t0 + t1 + t2 + t3 + t4) * 0.2f;
  }
  __syncthreads();
  if (tid < Q_LEN) {
    float x = lin_b[0];
#pragma unroll
    for (int i = 0; i < 6; ++i) x += featb[tid * 6 + i] * lin_w[i];
    lo[tid] = 1.f / (1.f + __expf(-x));
  }
  __syncthreads();
  if (tid == 0) {
    float s = 0.f;
    for (int q = 0; q < Q_LEN; ++q) s += lo[q];
    out[docSel * N_SENT + n] = s * (1.f / Q_LEN);
  }
}

// -------------------------------------------------------------------------------
extern "C" void kernel_launch(void* const* d_in, const int* in_sizes, int n_in,
                              void* d_out, int out_size, void* d_ws, size_t ws_size,
                              hipStream_t stream) {
  const float* embeds   = (const float*)d_in[0];
  const float* filters  = (const float*)d_in[1];
  const float* lin_w    = (const float*)d_in[2];
  const float* lin_b    = (const float*)d_in[3];
  const int*   question = (const int*)d_in[4];
  const int*   doc1     = (const int*)d_in[5];
  const int*   doc2     = (const int*)d_in[6];
  const float* sim1     = (const float*)d_in[7];
  const float* sim2     = (const float*)d_in[8];
  float* out = (float*)d_out;

  unsigned short* Wb  = (unsigned short*)d_ws;           // [FP_][FSZ][EP] bf16
  unsigned short* qbE = Wb  + (size_t)FP_ * FSZ * EP;    // [Q_LEN][EP]
  unsigned short* qbC = qbE + (size_t)Q_LEN * EP;        // [Q_LEN][EP]
  float* qnE = (float*)(qbC + (size_t)Q_LEN * EP);
  float* qnC = qnE + Q_LEN;

  int wElems = FP_ * FSZ * EP;
  prep_filters<<<(wElems + 255) / 256, 256, 0, stream>>>(filters, Wb);
  prep_question<<<1, 256, 0, stream>>>(embeds, filters, question, qbE, qbC, qnE, qnC);
  doc_kernel<<<dim3(N_SENT, 2), 256, 0, stream>>>(embeds, question, doc1, doc2, sim1, sim2,
                                                  Wb, qbE, qbC, qnE, qnC, lin_w, lin_b, out);
}
